// STAttModel_12670153523503
// MI455X (gfx1250) — matrix operand order repaired and checked by reading, not compiled, not run on hardware
//
#include <hip/hip_runtime.h>
#include <math.h>

// ---------------------------------------------------------------------------
// Problem constants (from reference)
// ---------------------------------------------------------------------------
#define BDIM   16
#define TDIM   12
#define NNODE  1000
#define DMODEL 64
#define KHEAD  8
#define DHEAD  8
#define MTOK   (BDIM * TDIM * NNODE)   // 192000 tokens
#define BT     (BDIM * TDIM)           // 192 spatial batches

// Use CDNA5 async global->LDS copies (ASYNCcnt-tracked) for GEMM tile staging.
#define USE_ASYNC_LDS 1

typedef _Float16 h16;
typedef __attribute__((ext_vector_type(16))) _Float16 v16h;
typedef __attribute__((ext_vector_type(8)))  float    v8f;
typedef __attribute__((address_space(3))) h16 lds_h16;

__device__ __forceinline__ float gelu_f(float x) {
    return 0.5f * x * (1.0f + erff(x * 0.70710678118654752f));
}

// ---------------------------------------------------------------------------
// Generic WMMA GEMM: C[M,64] = act( A[M,K] @ B[K,64] + bias (+resid) (+Cprev) )
//   * A can be a 2-segment concat along K (A0 for k<K0, A1 for k>=K0)
//   * batched via grid.y with per-operand element batch strides
//   * 256 threads = 8 waves; each wave owns a 16x64 slab (4 accumulators)
//   * K loop in steps of 64; double-buffered LDS tiles; next tile staged via
//     global_load_async_to_lds_b128 while 8 x v_wmma_f32_16x16x32_f16 run;
//     s_wait_asynccnt 0 + one barrier per K-step
// ---------------------------------------------------------------------------
__global__ __launch_bounds__(256) void wmma_gemm_kernel(
    const h16* __restrict__ A0, int lda0, int K0,
    const h16* __restrict__ A1, int lda1, long long bsA,
    const h16* __restrict__ Bw, int ldb, long long bsB,
    const float* __restrict__ bias,
    const float* __restrict__ resid, long long bsR,
    float* __restrict__ C32, long long bsC32, int beta,
    h16* __restrict__ C16, long long bsC16,
    int M, int Ktot, int act)
{
    __shared__ h16 As[2][128][64];   // 2 x 16 KB
    __shared__ h16 Bs[2][64][64];    // 2 x  8 KB

    const int tid  = threadIdx.x;
    const int lane = tid & 31;
    const int wave = tid >> 5;
    const int batch = blockIdx.y;
    const int rowBase = blockIdx.x * 128;

    if (A0)    A0    += (long long)batch * bsA;
    if (A1)    A1    += (long long)batch * bsA;
    Bw                += (long long)batch * bsB;
    if (C32)   C32   += (long long)batch * bsC32;
    if (C16)   C16   += (long long)batch * bsC16;
    if (resid) resid += (long long)batch * bsR;

    const bool interior = (rowBase + 128 <= M);

    // Stage a 128x64 A tile and 64x64 B tile (f16) into buffer `buf`.
    auto stage = [&](int k0, int buf) {
        if (interior && (k0 + 64 <= Ktot)) {
            // fast path: 16-byte chunks, pointer-select for the concat segment
            // (K0 is a multiple of 8 -> chunks never straddle segments)
            #pragma unroll
            for (int it = 0; it < 4; ++it) {
                int idx = tid + it * 256;            // 1024 chunks of 8 halves
                int r   = idx >> 3;                  // 0..127
                int kk  = (idx & 7) << 3;            // 0,8,..,56
                int gr  = rowBase + r;
                int gk  = k0 + kk;
                const h16* p = (gk < K0) ? (A0 + (long long)gr * lda0 + gk)
                                         : (A1 + (long long)gr * lda1 + (gk - K0));
#if USE_ASYNC_LDS
                lds_h16* d = (lds_h16*)&As[buf][r][kk];
                asm volatile("global_load_async_to_lds_b128 %0, %1, off"
                             :: "v"(d), "v"(p) : "memory");
#else
                *(uint4*)(&As[buf][r][kk]) = *(const uint4*)p;
#endif
            }
            #pragma unroll
            for (int it = 0; it < 2; ++it) {
                int idx = tid + it * 256;            // 512 chunks of 8 halves
                int kk  = idx >> 3;                  // 0..63
                int n   = (idx & 7) << 3;            // 0,8,..,56
                const h16* p = Bw + (long long)(k0 + kk) * ldb + n;
#if USE_ASYNC_LDS
                lds_h16* d = (lds_h16*)&Bs[buf][kk][n];
                asm volatile("global_load_async_to_lds_b128 %0, %1, off"
                             :: "v"(d), "v"(p) : "memory");
#else
                *(uint4*)(&Bs[buf][kk][n]) = *(const uint4*)p;
#endif
            }
        } else {
            // edge path: guarded scalar loads with zero fill
            #pragma unroll 1
            for (int idx = tid; idx < 128 * 64; idx += 256) {
                int r  = idx >> 6;
                int kk = idx & 63;
                int gr = rowBase + r, gk = k0 + kk;
                h16 val = (h16)0.0f;
                if (gr < M && gk < Ktot) {
                    const h16* p = (gk < K0) ? (A0 + (long long)gr * lda0 + gk)
                                             : (A1 + (long long)gr * lda1 + (gk - K0));
                    val = *p;
                }
                As[buf][r][kk] = val;
            }
            #pragma unroll 1
            for (int idx = tid; idx < 64 * 64; idx += 256) {
                int kk = idx >> 6;
                int n  = idx & 63;
                int gk = k0 + kk;
                Bs[buf][kk][n] = (gk < Ktot) ? Bw[(long long)gk * ldb + n] : (h16)0.0f;
            }
        }
    };

    // wait for this wave's outstanding async LDS copies, then publish via barrier
    auto publish = [&]() {
#if USE_ASYNC_LDS
        asm volatile("s_wait_asynccnt 0x0" ::: "memory");
#endif
        __syncthreads();
    };

    v8f acc0 = {}, acc1 = {}, acc2 = {}, acc3 = {};

    stage(0, 0);
    publish();

    int buf = 0;
    for (int k0 = 0; k0 < Ktot; k0 += 64) {
        // stage next K-step into the other buffer while this one is consumed
        if (k0 + 64 < Ktot) stage(k0 + 64, buf ^ 1);
        if (k0 + 128 < Ktot)
            __builtin_prefetch((const void*)(Bw + (long long)(k0 + 128) * ldb + ((tid & 7) << 3)), 0, 1);

        const int arow = wave * 16 + (lane & 15);
        const int kb   = (lane >> 4) << 3;                 // 0 or 8
        const int brow = (lane & 15) + ((lane >> 4) << 4); // K index within 32

        #pragma unroll
        for (int h = 0; h < 2; ++h) {                      // two 32-deep K halves
            const int kh = h * 32;
            v16h a;   // 16x32 f16 A fragment, layout per ISA 7.12.2
            #pragma unroll
            for (int e = 0; e < 8; ++e) a[e]     = As[buf][arow][kh + kb + e];
            #pragma unroll
            for (int e = 0; e < 8; ++e) a[e + 8] = As[buf][arow][kh + kb + 16 + e];
            {
                v16h b;
                #pragma unroll
                for (int e = 0; e < 16; ++e) b[e] = Bs[buf][kh + kb + e + ((e >> 3) << 3)][0 + (lane & 15)];
                acc0 = __builtin_amdgcn_wmma_f32_16x16x32_f16(false, a, false, b, (short)0, acc0, false, false);
            }
            {
                v16h b;
                #pragma unroll
                for (int e = 0; e < 16; ++e) b[e] = Bs[buf][kh + kb + e + ((e >> 3) << 3)][16 + (lane & 15)];
                acc1 = __builtin_amdgcn_wmma_f32_16x16x32_f16(false, a, false, b, (short)0, acc1, false, false);
            }
            {
                v16h b;
                #pragma unroll
                for (int e = 0; e < 16; ++e) b[e] = Bs[buf][kh + kb + e + ((e >> 3) << 3)][32 + (lane & 15)];
                acc2 = __builtin_amdgcn_wmma_f32_16x16x32_f16(false, a, false, b, (short)0, acc2, false, false);
            }
            {
                v16h b;
                #pragma unroll
                for (int e = 0; e < 16; ++e) b[e] = Bs[buf][kh + kb + e + ((e >> 3) << 3)][48 + (lane & 15)];
                acc3 = __builtin_amdgcn_wmma_f32_16x16x32_f16(false, a, false, b, (short)0, acc3, false, false);
            }
        }
        publish();
        buf ^= 1;
    }

    // ---- epilogue: C layout lane->(col = lane&15, rows = 8*(lane>>4)+i) ----
    const int rsub = wave * 16 + ((lane >> 4) << 3);
    const int ccol = lane & 15;
    v8f accs[4] = {acc0, acc1, acc2, acc3};
    #pragma unroll
    for (int t = 0; t < 4; ++t) {
        const int col = t * 16 + ccol;
        const float bv = bias ? bias[col] : 0.0f;
        #pragma unroll
        for (int i = 0; i < 8; ++i) {
            int grow = rowBase + rsub + i;
            if (grow >= M) continue;
            long long off = (long long)grow * 64 + col;
            float x = accs[t][i] + bv;
            if (resid) x += resid[off];
            if (beta && C32) x += C32[off];
            if (act == 1) x = gelu_f(x);
            if (C32) C32[off] = x;
            if (C16) C16[off] = (h16)x;
        }
    }
}

// ---------------------------------------------------------------------------
// Adjacency softmax for one GAT layer:
//   S[v,w] = softmax_w( adj[v,w]>0 ? (nv1[v,:] . nv2[:,w]) : -9e15 )  -> f16
// One block (256 threads, 8 wave32) per row v.
// ---------------------------------------------------------------------------
__global__ __launch_bounds__(256) void gat_softmax_kernel(
    const float* __restrict__ nv1, const float* __restrict__ nv2,
    const int* __restrict__ adj, h16* __restrict__ S, int N)
{
    __shared__ float row[1024];
    __shared__ float red[8];
    const int v = blockIdx.x;
    const int tid = threadIdx.x, lane = tid & 31, wave = tid >> 5;

    float a[10];
    #pragma unroll
    for (int j = 0; j < 10; ++j) a[j] = nv1[v * 10 + j];

    float lmax = -3.4e38f;
    for (int w = tid; w < N; w += 256) {
        float e = 0.0f;
        #pragma unroll
        for (int j = 0; j < 10; ++j) e += a[j] * nv2[j * N + w];
        e = (adj[(long long)v * N + w] > 0) ? e : -9.0e15f;
        row[w] = e;
        lmax = fmaxf(lmax, e);
    }
    #pragma unroll
    for (int off = 16; off > 0; off >>= 1) lmax = fmaxf(lmax, __shfl_down(lmax, off, 32));
    if (lane == 0) red[wave] = lmax;
    __syncthreads();
    float gmax = red[0];
    #pragma unroll
    for (int i = 1; i < 8; ++i) gmax = fmaxf(gmax, red[i]);
    __syncthreads();

    float lsum = 0.0f;
    for (int w = tid; w < N; w += 256) {
        float p = __expf(row[w] - gmax);
        row[w] = p;
        lsum += p;
    }
    #pragma unroll
    for (int off = 16; off > 0; off >>= 1) lsum += __shfl_down(lsum, off, 32);
    if (lane == 0) red[wave] = lsum;
    __syncthreads();
    float gsum = 0.0f;
    #pragma unroll
    for (int i = 0; i < 8; ++i) gsum += red[i];
    const float inv = 1.0f / gsum;
    for (int w = tid; w < N; w += 256)
        S[(long long)v * N + w] = (h16)(row[w] * inv);
}

// adpT[v,w] = (f16) adp[w,v]   (GCN uses A^T as the left GEMM operand)
__global__ void transpose_f16_kernel(const float* __restrict__ in, h16* __restrict__ out, int N)
{
    long long i = (long long)blockIdx.x * blockDim.x + threadIdx.x;
    if (i >= (long long)N * N) return;
    int r = (int)(i / N), c = (int)(i % N);
    out[i] = (h16)in[(long long)c * N + r];
}

__global__ void f32_to_f16_kernel(const float* __restrict__ in, h16* __restrict__ out, long long n)
{
    long long i = (long long)blockIdx.x * blockDim.x + threadIdx.x;
    if (i < n) out[i] = (h16)in[i];
}

// out32 = gelu(in) (in-place safe), out16 = f16(gelu(in))
__global__ void gelu_dual_kernel(const float* __restrict__ in, float* __restrict__ out32,
                                 h16* __restrict__ out16, long long n)
{
    long long i = (long long)blockIdx.x * blockDim.x + threadIdx.x;
    if (i >= n) return;
    float g = gelu_f(in[i]);
    out32[i] = g;
    out16[i] = (h16)g;
}

// Hf = sigmoid(Z) * HS + (1-sigmoid(Z)) * HT  -> f16
__global__ void gate_fuse_kernel(const float* __restrict__ Z, const float* __restrict__ HS,
                                 const float* __restrict__ HT, h16* __restrict__ out, long long n)
{
    long long i = (long long)blockIdx.x * blockDim.x + threadIdx.x;
    if (i >= n) return;
    float z = 1.0f / (1.0f + __expf(-Z[i]));
    out[i] = (h16)(z * HS[i] + (1.0f - z) * HT[i]);
}

// ---------------------------------------------------------------------------
// Causal temporal attention over T=12 per (head, b, n, t). q/k/v are f16
// [MTOK,64] with head h occupying columns h*8..h*8+7 (split_heads layout).
// ---------------------------------------------------------------------------
__global__ __launch_bounds__(256) void temporal_attn_kernel(
    const h16* __restrict__ q, const h16* __restrict__ k,
    const h16* __restrict__ v, h16* __restrict__ out)
{
    int id = blockIdx.x * 256 + threadIdx.x;
    const int total = KHEAD * BDIM * NNODE * TDIM;
    if (id >= total) return;
    int t = id % TDIM; int r = id / TDIM;
    int n = r % NNODE; r /= NNODE;
    int b = r % BDIM;  int kh = r / BDIM;

    const float scale = 0.35355339059327373f;  // 1/sqrt(8)
    long long qoff = ((long long)(b * TDIM + t) * NNODE + n) * DMODEL + kh * DHEAD;

    float qv[8];
    #pragma unroll
    for (int d = 0; d < 8; ++d) qv[d] = (float)q[qoff + d];

    float sc[TDIM];
    float mx = -3.4e38f;
    for (int s = 0; s <= t; ++s) {
        long long ko = ((long long)(b * TDIM + s) * NNODE + n) * DMODEL + kh * DHEAD;
        float e = 0.0f;
        #pragma unroll
        for (int d = 0; d < 8; ++d) e += qv[d] * (float)k[ko + d];
        sc[s] = e * scale;
        mx = fmaxf(mx, sc[s]);
    }
    float sum = 0.0f;
    for (int s = 0; s <= t; ++s) { sc[s] = __expf(sc[s] - mx); sum += sc[s]; }
    const float inv = 1.0f / sum;
    float o[8] = {0, 0, 0, 0, 0, 0, 0, 0};
    for (int s = 0; s <= t; ++s) {
        long long vo = ((long long)(b * TDIM + s) * NNODE + n) * DMODEL + kh * DHEAD;
        float w = sc[s] * inv;
        #pragma unroll
        for (int d = 0; d < 8; ++d) o[d] += w * (float)v[vo + d];
    }
    #pragma unroll
    for (int d = 0; d < 8; ++d) out[qoff + d] = (h16)o[d];
}

// ---------------------------------------------------------------------------
// Host-side orchestration
// ---------------------------------------------------------------------------
static void run_gemm(hipStream_t s,
                     const h16* A0, int lda0, int K0, const h16* A1, int lda1, long long bsA,
                     const h16* Bw, int ldb, long long bsB,
                     const float* bias, const float* resid, long long bsR,
                     float* C32, long long bsC32, int beta,
                     h16* C16, long long bsC16,
                     int M, int Ktot, int act, int nbatch)
{
    dim3 grid((M + 127) / 128, nbatch);
    wmma_gemm_kernel<<<grid, 256, 0, s>>>(A0, lda0, K0, A1, lda1, bsA, Bw, ldb, bsB,
                                          bias, resid, bsR, C32, bsC32, beta, C16, bsC16,
                                          M, Ktot, act);
}

extern "C" void kernel_launch(void* const* d_in, const int* in_sizes, int n_in,
                              void* d_out, int out_size, void* d_ws, size_t ws_size,
                              hipStream_t stream)
{
    (void)in_sizes; (void)n_in; (void)out_size; (void)ws_size;

    const float* X    = (const float*)d_in[0];
    const float* STE  = (const float*)d_in[1];
    const float* adp  = (const float*)d_in[2];
    const int*   adj0 = (const int*)  d_in[3];
    const int*   adj1 = (const int*)  d_in[4];
    const float* w30  = (const float*)d_in[5];
    const float* b30  = (const float*)d_in[6];
    const float* wg   = (const float*)d_in[7];
    const float* bg   = (const float*)d_in[8];
    const float* wgat[4] = {(const float*)d_in[9],  (const float*)d_in[13],
                            (const float*)d_in[17], (const float*)d_in[21]};
    const float* bgat[4] = {(const float*)d_in[10], (const float*)d_in[14],
                            (const float*)d_in[18], (const float*)d_in[22]};
    const float* nv1g[4] = {(const float*)d_in[11], (const float*)d_in[15],
                            (const float*)d_in[19], (const float*)d_in[23]};
    const float* nv2g[4] = {(const float*)d_in[12], (const float*)d_in[16],
                            (const float*)d_in[20], (const float*)d_in[24]};
    const float* w12 = (const float*)d_in[25]; const float* b12 = (const float*)d_in[26];
    const float* w13 = (const float*)d_in[27]; const float* b13 = (const float*)d_in[28];
    const float* w14 = (const float*)d_in[29]; const float* b14 = (const float*)d_in[30];
    const float* w15 = (const float*)d_in[31]; const float* b15 = (const float*)d_in[32];
    const float* w16 = (const float*)d_in[33]; const float* b16 = (const float*)d_in[34];
    const float* w17 = (const float*)d_in[35]; const float* b17 = (const float*)d_in[36];
    const float* w18 = (const float*)d_in[37]; const float* b18 = (const float*)d_in[38];
    const float* w19 = (const float*)d_in[39]; const float* b19 = (const float*)d_in[40];
    const float* w20 = (const float*)d_in[41]; const float* b20 = (const float*)d_in[42];

    // ---- workspace carve-out (256B aligned) ----
    char* wp = (char*)d_ws;
    auto walloc = [&](size_t bytes) -> void* {
        void* r = (void*)wp;
        wp += (bytes + 255) & ~(size_t)255;
        return r;
    };
    const long long ME = (long long)MTOK * DMODEL;  // 12,288,000 elements

    h16* Xh   = (h16*)walloc(ME * 2);   // X f16; reused as attention output later
    h16* STEh = (h16*)walloc(ME * 2);
    h16* T0   = (h16*)walloc(ME * 2);   // generic f16 temp
    h16* P0   = (h16*)walloc(ME * 2);   // ping
    h16* P1   = (h16*)walloc(ME * 2);   // pong
    h16* qh   = (h16*)walloc(ME * 2);
    h16* kh   = (h16*)walloc(ME * 2);
    h16* vh   = (h16*)walloc(ME * 2);
    h16* HSh  = (h16*)walloc(ME * 2);
    h16* HTh  = (h16*)walloc(ME * 2);
    float* Csum = (float*)walloc(ME * 4);   // HS pre-activation accumulator -> HS32
    float* HT32 = (float*)walloc(ME * 4);
    float* Zbuf = (float*)walloc(ME * 4);
    h16* Sh   = (h16*)walloc((long long)NNODE * NNODE * 2);  // softmaxed adjacency
    h16* SgT  = (h16*)walloc((long long)NNODE * NNODE * 2);  // adp^T for GCN
    h16* w30h = (h16*)walloc(448 * 64 * 2);
    h16* wgh  = (h16*)walloc(128 * 64 * 2);
    h16* wgath[4];
    for (int i = 0; i < 4; ++i) wgath[i] = (h16*)walloc(128 * 64 * 2);
    h16* w12h = (h16*)walloc(128 * 64 * 2);
    h16* w13h = (h16*)walloc(128 * 64 * 2);
    h16* w14h = (h16*)walloc(128 * 64 * 2);
    h16* w15h = (h16*)walloc(64 * 64 * 2);
    h16* w16h = (h16*)walloc(64 * 64 * 2);
    h16* w17h = (h16*)walloc(64 * 64 * 2);
    h16* w18h = (h16*)walloc(64 * 64 * 2);
    h16* w19h = (h16*)walloc(64 * 64 * 2);
    h16* w20h = (h16*)walloc(64 * 64 * 2);

    auto cvt = [&](const float* src, h16* dst, long long n) {
        f32_to_f16_kernel<<<dim3((unsigned)((n + 255) / 256)), 256, 0, stream>>>(src, dst, n);
    };

    // ---- precision staging ----
    cvt(X, Xh, ME);
    cvt(STE, STEh, ME);
    cvt(w30, w30h, 448 * 64);
    cvt(wg, wgh, 128 * 64);
    for (int i = 0; i < 4; ++i) cvt(wgat[i], wgath[i], 128 * 64);
    cvt(w12, w12h, 128 * 64); cvt(w13, w13h, 128 * 64); cvt(w14, w14h, 128 * 64);
    cvt(w15, w15h, 64 * 64);  cvt(w16, w16h, 64 * 64);  cvt(w17, w17h, 64 * 64);
    cvt(w18, w18h, 64 * 64);  cvt(w19, w19h, 64 * 64);  cvt(w20, w20h, 64 * 64);
    transpose_f16_kernel<<<dim3((NNODE * NNODE + 255) / 256), 256, 0, stream>>>(adp, SgT, NNODE);

    const long long bsTile = (long long)NNODE * DMODEL;  // per-(b,t) stride

    // helper lambdas for the three GEMM shapes
    auto linear_cat = [&](const h16* Ain, const h16* Wh, const float* bias, h16* out16) {
        // gelu([Ain|STE] @ W + b), M=MTOK, K=128
        run_gemm(stream, Ain, 64, 64, STEh, 64, 0, Wh, 64, 0, bias, nullptr, 0,
                 nullptr, 0, 0, out16, 0, MTOK, 128, /*gelu*/1, 1);
    };
    auto spatial = [&](const h16* Smat, const h16* Hsrc, h16* out16, int act) {
        // per (b,t): Smat[1000,1000] @ H[1000,64]
        run_gemm(stream, Smat, NNODE, NNODE, nullptr, 0, 0, Hsrc, 64, bsTile,
                 nullptr, nullptr, 0, nullptr, 0, 0, out16, bsTile, NNODE, NNODE, act, BT);
    };
    auto hs_acc = [&](const h16* seg, int j, const float* bias, int beta) {
        // Csum += seg @ w30[j*64:(j+1)*64, :]
        run_gemm(stream, seg, 64, 64, nullptr, 0, 0, w30h + (long long)j * 64 * 64, 64, 0,
                 bias, nullptr, 0, Csum, 0, beta, nullptr, 0, MTOK, 64, 0, 1);
    };

    // ---- spatial branch: 4 GAT + 2 GCN, accumulated into Csum ----
    hs_acc(Xh, 0, b30, /*beta=*/0);                       // X segment (with bias)

    // GAT1 (adj0): H=gelu(lin(X,STE)); HS1=gelu(S1@H) -> P0
    gat_softmax_kernel<<<NNODE, 256, 0, stream>>>(nv1g[0], nv2g[0], adj0, Sh, NNODE);
    linear_cat(Xh, wgath[0], bgat[0], T0);
    spatial(Sh, T0, P0, 1);
    hs_acc(P0, 1, nullptr, 1);

    // GAT2 (adj0, input HS1): -> P1
    gat_softmax_kernel<<<NNODE, 256, 0, stream>>>(nv1g[1], nv2g[1], adj0, Sh, NNODE);
    linear_cat(P0, wgath[1], bgat[1], T0);
    spatial(Sh, T0, P1, 1);
    hs_acc(P1, 2, nullptr, 1);

    // GAT3 (adj1, input X): -> P0
    gat_softmax_kernel<<<NNODE, 256, 0, stream>>>(nv1g[2], nv2g[2], adj1, Sh, NNODE);
    linear_cat(Xh, wgath[2], bgat[2], T0);
    spatial(Sh, T0, P0, 1);
    hs_acc(P0, 3, nullptr, 1);

    // GAT4 (adj1, input HS3): -> P1
    gat_softmax_kernel<<<NNODE, 256, 0, stream>>>(nv1g[3], nv2g[3], adj1, Sh, NNODE);
    linear_cat(P0, wgath[3], bgat[3], T0);
    spatial(Sh, T0, P1, 1);
    hs_acc(P1, 4, nullptr, 1);

    // GCN5 (input X): HS5 = adp^T @ gelu(lin(X,STE))  (no outer gelu) -> P0
    linear_cat(Xh, wgh, bg, T0);
    spatial(SgT, T0, P0, 0);
    hs_acc(P0, 5, nullptr, 1);

    // GCN6 (input HS5): -> P1
    linear_cat(P0, wgh, bg, T0);
    spatial(SgT, T0, P1, 0);
    hs_acc(P1, 6, nullptr, 1);

    // HS = gelu(Csum)  (f32 in place + f16 copy)
    gelu_dual_kernel<<<dim3((unsigned)((ME + 255) / 256)), 256, 0, stream>>>(Csum, Csum, HSh, ME);

    // ---- temporal branch ----
    linear_cat(Xh, w12h, b12, qh);
    linear_cat(Xh, w13h, b13, kh);
    linear_cat(Xh, w14h, b14, vh);
    {
        const int total = KHEAD * BDIM * NNODE * TDIM;
        temporal_attn_kernel<<<dim3((total + 255) / 256), 256, 0, stream>>>(qh, kh, vh, Xh); // Xh reused
    }
    // mid = gelu(attn@w15+b15) -> T0 ; HT = mid@w16+b16 -> HT32/HTh
    run_gemm(stream, Xh, 64, 64, nullptr, 0, 0, w15h, 64, 0, b15, nullptr, 0,
             nullptr, 0, 0, T0, 0, MTOK, 64, 1, 1);
    run_gemm(stream, T0, 64, 64, nullptr, 0, 0, w16h, 64, 0, b16, nullptr, 0,
             HT32, 0, 0, HTh, 0, MTOK, 64, 0, 1);

    // ---- gated fusion ----
    run_gemm(stream, HSh, 64, 64, nullptr, 0, 0, w17h, 64, 0, b17, nullptr, 0,
             Zbuf, 0, 0, nullptr, 0, MTOK, 64, 0, 1);
    run_gemm(stream, HTh, 64, 64, nullptr, 0, 0, w18h, 64, 0, b18, nullptr, 0,
             Zbuf, 0, 1, nullptr, 0, MTOK, 64, 0, 1);
    gate_fuse_kernel<<<dim3((unsigned)((ME + 255) / 256)), 256, 0, stream>>>(Zbuf, Csum, HT32, T0, ME);

    // out = X + (gelu(Hf@w19+b19) @ w20 + b20)
    run_gemm(stream, T0, 64, 64, nullptr, 0, 0, w19h, 64, 0, b19, nullptr, 0,
             nullptr, 0, 0, P0, 0, MTOK, 64, 1, 1);
    run_gemm(stream, P0, 64, 64, nullptr, 0, 0, w20h, 64, 0, b20, X, 0,
             (float*)d_out, 0, 0, nullptr, 0, MTOK, 64, 0, 1);
}
